// SelfAttention_9096740733116
// MI455X (gfx1250) — matrix operand order repaired and checked by reading, not compiled, not run on hardware
//
#include <hip/hip_runtime.h>

// Problem constants (from reference): b=4, s=2048, d=1024, 16 heads, dh=64.
#define BB 4
#define SS 2048
#define DD 1024
#define HH 16
#define DHD 64

typedef __attribute__((ext_vector_type(16))) __bf16 v16bf;
typedef __attribute__((ext_vector_type(8)))  float v8f;

// ---------------- helpers ----------------

static __device__ __forceinline__ unsigned short f2bf(float f) {
  unsigned u = __builtin_bit_cast(unsigned, f);
  u += 0x7FFFu + ((u >> 16) & 1u);          // round-to-nearest-even
  return (unsigned short)(u >> 16);
}

union FragU { v16bf v; uint4 q[2]; };

// A-operand fragment (16x32 bf16), ISA 7.12.2 layout:
// lane half hf: elems 0..7 = K(k0 + hf*8 .. +7), elems 8..15 = K(k0+16+hf*8 .. +7)
static __device__ __forceinline__ v16bf load_afrag(const unsigned short* base, int ld,
                                                   int m, int hf, int k0) {
  const unsigned short* p = base + (size_t)m * ld + k0 + hf * 8;
  FragU u;
  u.q[0] = *(const uint4*)(p);
  u.q[1] = *(const uint4*)(p + 16);
  return u.v;
}

// B-operand fragment (32x16 bf16) loaded from B^T row-major (row = n, stride ld):
// lane half hf: elems 0..7 = K(k0 + hf*8 .. +7), elems 8..15 = K(k0+16+hf*8 .. +7)
static __device__ __forceinline__ v16bf load_bfrag(const unsigned short* base, int ld,
                                                   int n, int hf, int k0) {
  const unsigned short* p = base + (size_t)n * ld + k0 + hf * 8;
  FragU u;
  u.q[0] = *(const uint4*)(p);
  u.q[1] = *(const uint4*)(p + 16);
  return u.v;
}

static __device__ __forceinline__ v8f wmma_bf16(v16bf a, v16bf b, v8f c) {
  return __builtin_amdgcn_wmma_f32_16x16x32_bf16(false, a, false, b, (short)0, c, false, false);
}

static __device__ __forceinline__ float redmax16(float v) {
#pragma unroll
  for (int off = 1; off < 16; off <<= 1) v = fmaxf(v, __shfl_xor(v, off, 32));
  return v;
}
static __device__ __forceinline__ float redsum16(float v) {
#pragma unroll
  for (int off = 1; off < 16; off <<= 1) v += __shfl_xor(v, off, 32);
  return v;
}

// ---------------- preprocessing ----------------

__global__ void cvt_x_kernel(const float* __restrict__ x, unsigned short* __restrict__ xb, int n) {
  int i = (blockIdx.x * blockDim.x + threadIdx.x) * 4;
  if (i + 3 < n) {
    float4 v = *(const float4*)(x + i);
    xb[i + 0] = f2bf(v.x);
    xb[i + 1] = f2bf(v.y);
    xb[i + 2] = f2bf(v.z);
    xb[i + 3] = f2bf(v.w);
  }
}

// Wt[z][n][k] = bf16(W[k][n])  for z in {q,k,v,o}
__global__ void transpose_w_kernel(const float* __restrict__ Wq, const float* __restrict__ Wk,
                                   const float* __restrict__ Wv, const float* __restrict__ Wo,
                                   unsigned short* __restrict__ wt) {
  __shared__ float tile[32][33];
  const float* src = (blockIdx.z == 0) ? Wq : (blockIdx.z == 1) ? Wk : (blockIdx.z == 2) ? Wv : Wo;
  unsigned short* dst = wt + (size_t)blockIdx.z * DD * DD;
  int k0 = blockIdx.x * 32, n0 = blockIdx.y * 32;
  int tx = threadIdx.x, ty = threadIdx.y;   // block (32, 8)
#pragma unroll
  for (int i = 0; i < 4; ++i)
    tile[ty + i * 8][tx] = src[(size_t)(k0 + ty + i * 8) * DD + n0 + tx];
  __syncthreads();
#pragma unroll
  for (int i = 0; i < 4; ++i)
    dst[(size_t)(n0 + ty + i * 8) * DD + k0 + tx] = f2bf(tile[tx][ty + i * 8]);
}

// ---------------- fused QKV projection ----------------
// grid: (rows/64, DD/64, 3); block 128 (4 waves); wave computes a 16x64 tile.
// K-loop is ping-pong double-buffered (2x unroll): set B loads while set A feeds
// WMMAs and vice versa -- no register rotation copies, no WAR hazard NOPs.
__global__ __launch_bounds__(128) void qkv_kernel(
    const unsigned short* __restrict__ xb, const unsigned short* __restrict__ wt,
    const float* __restrict__ bq, const float* __restrict__ bk, const float* __restrict__ bv,
    unsigned short* __restrict__ Qb, unsigned short* __restrict__ Kb, unsigned short* __restrict__ Vt) {
  int wave = threadIdx.x >> 5, lane = threadIdx.x & 31;
  int m = lane & 15, hf = lane >> 4;
  int row0 = (blockIdx.x * 4 + wave) * 16;
  int n0 = blockIdx.y * 64;
  int z = blockIdx.z;
  const unsigned short* wz = wt + (size_t)z * DD * DD;
  const float* bias = (z == 0) ? bq : (z == 1) ? bk : bv;
  const unsigned short* arow = xb + (size_t)row0 * DD;
  const unsigned short* wrow[4];
#pragma unroll
  for (int t = 0; t < 4; ++t) wrow[t] = wz + (size_t)(n0 + t * 16) * DD;

  v8f acc[4] = {};
  v16bf aA = load_afrag(arow, DD, m, hf, 0);
  v16bf bA[4];
#pragma unroll
  for (int t = 0; t < 4; ++t) bA[t] = load_bfrag(wrow[t], DD, m, hf, 0);
  v16bf aB, bB[4];

  for (int k0 = 0; k0 < DD - 64; k0 += 64) {
    aB = load_afrag(arow, DD, m, hf, k0 + 32);
#pragma unroll
    for (int t = 0; t < 4; ++t) bB[t] = load_bfrag(wrow[t], DD, m, hf, k0 + 32);
    __builtin_prefetch(arow + (size_t)m * DD + k0 + 96, 0, 0);  // global_prefetch_b8
#pragma unroll
    for (int t = 0; t < 4; ++t) acc[t] = wmma_bf16(aA, bA[t], acc[t]);
    aA = load_afrag(arow, DD, m, hf, k0 + 64);
#pragma unroll
    for (int t = 0; t < 4; ++t) bA[t] = load_bfrag(wrow[t], DD, m, hf, k0 + 64);
#pragma unroll
    for (int t = 0; t < 4; ++t) acc[t] = wmma_bf16(aB, bB[t], acc[t]);
  }
  // tail: last 64 of K
  aB = load_afrag(arow, DD, m, hf, DD - 32);
#pragma unroll
  for (int t = 0; t < 4; ++t) bB[t] = load_bfrag(wrow[t], DD, m, hf, DD - 32);
#pragma unroll
  for (int t = 0; t < 4; ++t) acc[t] = wmma_bf16(aA, bA[t], acc[t]);
#pragma unroll
  for (int t = 0; t < 4; ++t) acc[t] = wmma_bf16(aB, bB[t], acc[t]);

#pragma unroll
  for (int t = 0; t < 4; ++t) {
    int col = n0 + t * 16 + m;
    float bi = bias[col];
    int head = col >> 6, dhi = col & (DHD - 1);
#pragma unroll
    for (int r = 0; r < 8; ++r) {
      int row = row0 + r + 8 * hf;
      int bidx = row >> 11, sq = row & (SS - 1);
      unsigned short val = f2bf(acc[t][r] + bi);
      if (z == 2)  // V stored transposed: [b, h, dh, s]
        Vt[((size_t)(bidx * HH + head) * DHD + dhi) * SS + sq] = val;
      else {       // Q, K stored [b, h, s, dh]
        unsigned short* dst = (z == 0) ? Qb : Kb;
        dst[((size_t)(bidx * HH + head) * SS + sq) * DHD + dhi] = val;
      }
    }
  }
}

// ---------------- flash attention ----------------
// grid: (BB*HH, SS/64); block 128; wave owns 16 queries; online softmax over 32-key blocks.
// V-frags load up front so the softmax VALU section hides their latency; next K/V
// blocks are prefetched during compute.
__global__ __launch_bounds__(128) void attn_kernel(
    const unsigned short* __restrict__ Qb, const unsigned short* __restrict__ Kb,
    const unsigned short* __restrict__ Vt, const int* __restrict__ maskp,
    unsigned short* __restrict__ Ab) {
  int wave = threadIdx.x >> 5, lane = threadIdx.x & 31;
  int m = lane & 15, hf = lane >> 4;
  int bh = blockIdx.x;
  int m0 = (blockIdx.y * 4 + wave) * 16;
  int causal = maskp[0] != 0;

  const unsigned short* Qp = Qb + (size_t)bh * SS * DHD;
  const unsigned short* Kp = Kb + (size_t)bh * SS * DHD;
  const unsigned short* Vp = Vt + (size_t)bh * DHD * SS;

  v16bf q0 = load_afrag(Qp + (size_t)m0 * DHD, DHD, m, hf, 0);
  v16bf q1 = load_afrag(Qp + (size_t)m0 * DHD, DHD, m, hf, 32);

  v8f O[4] = {};
  float mrow[8], lrow[8];
#pragma unroll
  for (int r = 0; r < 8; ++r) { mrow[r] = -INFINITY; lrow[r] = 0.0f; }

  __shared__ unsigned short Ptile[4][16 * 40];   // per-wave 16x32 bf16 tile, padded stride 40
  unsigned short* myP = &Ptile[wave][0];

  const float scale = 0.125f;  // 1/sqrt(64)
  int jend = causal ? (m0 + 16) : SS;
  for (int j0 = 0; j0 < jend; j0 += 32) {
    // ---- issue all loads for this block early ----
    v16bf kb0 = load_bfrag(Kp + (size_t)j0 * DHD, DHD, m, hf, 0);
    v16bf kb1 = load_bfrag(Kp + (size_t)j0 * DHD, DHD, m, hf, 32);
    v16bf kb2 = load_bfrag(Kp + (size_t)(j0 + 16) * DHD, DHD, m, hf, 0);
    v16bf kb3 = load_bfrag(Kp + (size_t)(j0 + 16) * DHD, DHD, m, hf, 32);
    v16bf vb[4];
#pragma unroll
    for (int t = 0; t < 4; ++t) vb[t] = load_bfrag(Vp + (size_t)(t * 16) * SS, SS, m, hf, j0);
    if (j0 + 32 < jend) {  // prefetch next key/value blocks (128B lines per lane)
      __builtin_prefetch(Kp + (size_t)(j0 + 32 + m) * DHD, 0, 0);
      __builtin_prefetch(Kp + (size_t)(j0 + 48 + m) * DHD, 0, 0);
#pragma unroll
      for (int t = 0; t < 4; ++t)
        __builtin_prefetch(Vp + (size_t)(t * 16 + m) * SS + j0 + 32, 0, 0);
    }

    // ---- scores: S[16q x 32k] via 4 WMMAs (K over dh=64) ----
    v8f S0 = {}, S1 = {};
    S0 = wmma_bf16(q0, kb0, S0);
    S0 = wmma_bf16(q1, kb1, S0);
    S1 = wmma_bf16(q0, kb2, S1);
    S1 = wmma_bf16(q1, kb3, S1);

    // ---- online softmax update per row (row = r + 8*hf of this q-tile) ----
#pragma unroll
    for (int r = 0; r < 8; ++r) {
      int row = m0 + r + 8 * hf;
      int c0 = j0 + m, c1 = j0 + 16 + m;
      float s0 = S0[r] * scale;
      float s1 = S1[r] * scale;
      if (causal && c0 > row) s0 = -INFINITY;
      if (causal && c1 > row) s1 = -INFINITY;
      float bmax = redmax16(fmaxf(s0, s1));
      float mn = fmaxf(mrow[r], bmax);
      float p0 = __expf(s0 - mn);
      float p1 = __expf(s1 - mn);
      float bsum = redsum16(p0 + p1);
      float alpha = __expf(mrow[r] - mn);   // first block: exp(-inf)=0
      lrow[r] = lrow[r] * alpha + bsum;
      mrow[r] = mn;
#pragma unroll
      for (int t = 0; t < 4; ++t) O[t][r] *= alpha;
      myP[(r + 8 * hf) * 40 + m] = f2bf(p0);
      myP[(r + 8 * hf) * 40 + 16 + m] = f2bf(p1);
    }
    asm volatile("s_wait_dscnt 0" ::: "memory");  // intra-wave LDS store->load ordering

    // ---- P (A-frag from LDS) x V^T (B-frag): accumulate O[16q x 64dh] ----
    v16bf pf = load_afrag(myP, 40, m, hf, 0);
#pragma unroll
    for (int t = 0; t < 4; ++t) O[t] = wmma_bf16(pf, vb[t], O[t]);
  }

  // ---- normalize and write attn-out bf16 [b, s, h*dh] ----
  int bidx = bh >> 4, head = bh & (HH - 1);
#pragma unroll
  for (int t = 0; t < 4; ++t) {
#pragma unroll
    for (int r = 0; r < 8; ++r) {
      int row = m0 + r + 8 * hf;
      float o = O[t][r] / lrow[r];
      Ab[((size_t)bidx * SS + row) * DD + head * DHD + t * 16 + m] = f2bf(o);
    }
  }
}

// ---------------- output projection ----------------
__global__ __launch_bounds__(128) void proj_kernel(
    const unsigned short* __restrict__ Ab, const unsigned short* __restrict__ woT,
    const float* __restrict__ bo, float* __restrict__ out) {
  int wave = threadIdx.x >> 5, lane = threadIdx.x & 31;
  int m = lane & 15, hf = lane >> 4;
  int row0 = (blockIdx.x * 4 + wave) * 16;
  int n0 = blockIdx.y * 64;
  const unsigned short* arow = Ab + (size_t)row0 * DD;
  const unsigned short* wrow[4];
#pragma unroll
  for (int t = 0; t < 4; ++t) wrow[t] = woT + (size_t)(n0 + t * 16) * DD;

  v8f acc[4] = {};
  v16bf aA = load_afrag(arow, DD, m, hf, 0);
  v16bf bA[4];
#pragma unroll
  for (int t = 0; t < 4; ++t) bA[t] = load_bfrag(wrow[t], DD, m, hf, 0);
  v16bf aB, bB[4];

  for (int k0 = 0; k0 < DD - 64; k0 += 64) {
    aB = load_afrag(arow, DD, m, hf, k0 + 32);
#pragma unroll
    for (int t = 0; t < 4; ++t) bB[t] = load_bfrag(wrow[t], DD, m, hf, k0 + 32);
    __builtin_prefetch(arow + (size_t)m * DD + k0 + 96, 0, 0);
#pragma unroll
    for (int t = 0; t < 4; ++t) acc[t] = wmma_bf16(aA, bA[t], acc[t]);
    aA = load_afrag(arow, DD, m, hf, k0 + 64);
#pragma unroll
    for (int t = 0; t < 4; ++t) bA[t] = load_bfrag(wrow[t], DD, m, hf, k0 + 64);
#pragma unroll
    for (int t = 0; t < 4; ++t) acc[t] = wmma_bf16(aB, bB[t], acc[t]);
  }
  aB = load_afrag(arow, DD, m, hf, DD - 32);
#pragma unroll
  for (int t = 0; t < 4; ++t) bB[t] = load_bfrag(wrow[t], DD, m, hf, DD - 32);
#pragma unroll
  for (int t = 0; t < 4; ++t) acc[t] = wmma_bf16(aA, bA[t], acc[t]);
#pragma unroll
  for (int t = 0; t < 4; ++t) acc[t] = wmma_bf16(aB, bB[t], acc[t]);

#pragma unroll
  for (int t = 0; t < 4; ++t) {
    int col = n0 + t * 16 + m;
    float bi = bo[col];
#pragma unroll
    for (int r = 0; r < 8; ++r) {
      int row = row0 + r + 8 * hf;
      out[(size_t)row * DD + col] = acc[t][r] + bi;
    }
  }
}

// ---------------- launch ----------------
extern "C" void kernel_launch(void* const* d_in, const int* in_sizes, int n_in,
                              void* d_out, int out_size, void* d_ws, size_t ws_size,
                              hipStream_t stream) {
  const float* x  = (const float*)d_in[0];
  const float* Wq = (const float*)d_in[1];
  const float* bq = (const float*)d_in[2];
  const float* Wk = (const float*)d_in[3];
  const float* bk = (const float*)d_in[4];
  const float* Wv = (const float*)d_in[5];
  const float* bv = (const float*)d_in[6];
  const float* Wo = (const float*)d_in[7];
  const float* bo = (const float*)d_in[8];
  const int* mask = (const int*)d_in[9];

  const size_t xdN = (size_t)BB * SS * DD;
  char* ws = (char*)d_ws;
  size_t off = 0;
  unsigned short* xb = (unsigned short*)(ws + off); off += xdN * 2;
  unsigned short* wt = (unsigned short*)(ws + off); off += (size_t)4 * DD * DD * 2;
  unsigned short* Qb = (unsigned short*)(ws + off); off += xdN * 2;
  unsigned short* Kb = (unsigned short*)(ws + off); off += xdN * 2;
  unsigned short* Vt = (unsigned short*)(ws + off); off += xdN * 2;
  unsigned short* Ab = (unsigned short*)(ws + off); off += xdN * 2;
  if (off > ws_size) return;  // workspace too small; nothing safe to do

  cvt_x_kernel<<<(int)(xdN / 1024), 256, 0, stream>>>(x, xb, (int)xdN);
  transpose_w_kernel<<<dim3(DD / 32, DD / 32, 4), dim3(32, 8), 0, stream>>>(Wq, Wk, Wv, Wo, wt);
  qkv_kernel<<<dim3(BB * SS / 64, DD / 64, 3), 128, 0, stream>>>(xb, wt, bq, bk, bv, Qb, Kb, Vt);
  attn_kernel<<<dim3(BB * HH, SS / 64), 128, 0, stream>>>(Qb, Kb, Vt, mask, Ab);
  proj_kernel<<<dim3(BB * SS / 64, DD / 64), 128, 0, stream>>>(Ab, wt + (size_t)3 * DD * DD, bo, (float*)d_out);
}
